// VQLinearPackSIMT_6030134083830
// MI455X (gfx1250) — matrix operand order repaired and checked by reading, not compiled, not run on hardware
//
#include <hip/hip_runtime.h>

// ---------------------------------------------------------------------------
// VQ-packed linear: out(512x11008) = X(512x4096) @ dequant(qweight, lut)^T
// CDNA5 (gfx1250, wave32) WMMA f16 GEMM with byte-pair LUT dequantization.
// ---------------------------------------------------------------------------

typedef __attribute__((ext_vector_type(16))) _Float16 v16h;
typedef __attribute__((ext_vector_type(8)))  _Float16 v8h;
typedef __attribute__((ext_vector_type(2)))  __fp16   v2fp16;   // cvt_pkrtz return type
typedef __attribute__((ext_vector_type(8)))  float    v8f;

#define IN_FEATURES   4096
#define OUT_FEATURES  11008
#define PACK_COLS     256      // int32 words per output row
#define BATCH         512      // 8*64 flattened rows

#define KSTEP   32                          // K per pipeline stage
#define NSTEPS  (IN_FEATURES / KSTEP)       // 128
#define TILE_B  64                          // batch rows per workgroup
#define TILE_M  256                         // out-features per workgroup (43*256=11008)
#define XPAD    40                          // f16 per LDS row (32 + 8 pad)
#define WPAD    40

union Frag  { v16h v; v8h h[2]; };
union Pack2 { v2fp16 v; unsigned u; };

static __device__ __forceinline__ unsigned pk(float lo, float hi) {
    Pack2 p; p.v = __builtin_amdgcn_cvt_pkrtz(lo, hi);   // v_cvt_pk_rtz_f16_f32
    return p.u;
}

__global__ __launch_bounds__(256)
void vq_wmma_gemm(const float* __restrict__ x,
                  const int*   __restrict__ qw,
                  const float* __restrict__ lut,
                  float*       __restrict__ out)
{
    __shared__ _Float16 sX[2][TILE_B * XPAD];   // X tile, f16, [row][k]
    __shared__ _Float16 sW[2][TILE_M * WPAD];   // W tile, f16, [mrow][k]
    __shared__ uint2    sLut2[256];             // byte (2 codes) -> 4 packed f16

    const int tid  = threadIdx.x;
    const int lane = tid & 31;
    const int wid  = tid >> 5;         // 8 waves
    const int wb   = wid >> 2;         // 0..1 : batch quadrant (32 rows)
    const int wn   = wid & 3;          // 0..3 : outfeat quadrant (64 cols)
    const int m0   = blockIdx.x * TILE_M;
    const int b0   = blockIdx.y * TILE_B;

    // Expand LUT to byte-pair table: entry c = {pack(lut[c&15]), pack(lut[c>>4])}
    // (one 8-bit lookup yields 4 consecutive K weights as 4 f16).
    {
        const int lo = tid & 15, hi = tid >> 4;
        sLut2[tid] = make_uint2(pk(lut[lo * 2], lut[lo * 2 + 1]),
                                pk(lut[hi * 2], lut[hi * 2 + 1]));
    }
    __syncthreads();

    // Every thread stages one W row (2 qweight words) and 8 X floats per step.
    const int wrow = tid;              // 0..255 outfeat row within tile
    const int urow = tid >> 2;         // 0..63  batch row within tile
    const int uqt  = tid & 3;          // which 8-float chunk of the K step

    const uint2*  qwp = (const uint2*)(qw + (long)(m0 + wrow) * PACK_COLS);
    const float4* xp  = (const float4*)(x + (long)(b0 + urow) * IN_FEATURES) + uqt * 2;

    uint2  wreg;
    float4 xr0, xr1;

    // Prefetch stage 0 from global into registers.
    wreg = qwp[0];
    xr0 = xp[0]; xr1 = xp[1];

    v8f acc[2][4] = {};                // [batch 16-tile][outfeat 16-tile]

    const int half = lane >> 4;
    const int l15  = lane & 15;

    for (int ks = 0; ks < NSTEPS; ++ks) {
        const int buf = ks & 1;

        // ---- stage registers -> LDS ----
        {   // W dequant: 8 byte lookups -> 32 f16 K-values for this row.
            unsigned* dst = (unsigned*)&sW[buf][wrow * WPAD];
            unsigned t[16];
            #pragma unroll
            for (int j = 0; j < 4; ++j) {
                uint2 e = sLut2[(wreg.x >> (8 * j)) & 255];
                t[2 * j + 0] = e.x; t[2 * j + 1] = e.y;
            }
            #pragma unroll
            for (int j = 0; j < 4; ++j) {
                uint2 e = sLut2[(wreg.y >> (8 * j)) & 255];
                t[8 + 2 * j + 0] = e.x; t[8 + 2 * j + 1] = e.y;
            }
            #pragma unroll
            for (int j = 0; j < 4; ++j)
                ((uint4*)dst)[j] = make_uint4(t[4*j], t[4*j+1], t[4*j+2], t[4*j+3]);
        }
        {   // X convert: 8 f32 -> 8 f16 via v_cvt_pk_rtz_f16_f32.
            unsigned* dst = (unsigned*)&sX[buf][urow * XPAD + uqt * 8];
            ((uint4*)dst)[0] = make_uint4(pk(xr0.x, xr0.y), pk(xr0.z, xr0.w),
                                          pk(xr1.x, xr1.y), pk(xr1.z, xr1.w));
        }
        __syncthreads();

        // ---- prefetch next stage from global ----
        if (ks + 1 < NSTEPS) {
            wreg = qwp[ks + 1];
            const float4* p = xp + (ks + 1) * (KSTEP / 4);
            xr0 = p[0]; xr1 = p[1];
        }

        // ---- build fragments from LDS and issue WMMA ----
        // A (16x32 f16): lane = M row; V0-3 hold K = half*8 + {0..7},
        //                V4-7 hold K = 16 + half*8 + {0..7}  (ISA 7.12.2)
        Frag a[2], bf[4];
        #pragma unroll
        for (int mt = 0; mt < 2; ++mt) {
            const _Float16* rb = &sX[buf][(wb * 32 + mt * 16 + l15) * XPAD];
            a[mt].h[0] = *(const v8h*)(rb + half * 8);
            a[mt].h[1] = *(const v8h*)(rb + 16 + half * 8);
        }
        // B (32x16 f16): lane = N col; V0-3 hold K = half*8 + {0..7},
        //                V4-7 hold K = 16 + half*8 + {0..7}
        #pragma unroll
        for (int nt = 0; nt < 4; ++nt) {
            const _Float16* rb = &sW[buf][(wn * 64 + nt * 16 + l15) * WPAD];
            bf[nt].h[0] = *(const v8h*)(rb + half * 8);
            bf[nt].h[1] = *(const v8h*)(rb + 16 + half * 8);
        }

        #pragma unroll
        for (int mt = 0; mt < 2; ++mt)
            #pragma unroll
            for (int nt = 0; nt < 4; ++nt)
                acc[mt][nt] = __builtin_amdgcn_wmma_f32_16x16x32_f16(
                    false, a[mt].v, false, bf[nt].v,
                    (short)0, acc[mt][nt], false, false);
        // no barrier needed here: next iteration writes the other LDS buffer,
        // and its barrier orders those writes against this compute.
    }

    // ---- epilogue: D layout -> out[b, m] (f32) ----
    // D: lane l15 = N (outfeature), VGPR r = M row (+8 for upper half-wave)
    #pragma unroll
    for (int mt = 0; mt < 2; ++mt) {
        #pragma unroll
        for (int nt = 0; nt < 4; ++nt) {
            const int bb = b0 + wb * 32 + mt * 16 + half * 8;
            const int mm = m0 + wn * 64 + nt * 16 + l15;
            #pragma unroll
            for (int r = 0; r < 8; ++r)
                out[(long)(bb + r) * OUT_FEATURES + mm] = acc[mt][nt][r];
        }
    }
}

extern "C" void kernel_launch(void* const* d_in, const int* in_sizes, int n_in,
                              void* d_out, int out_size, void* d_ws, size_t ws_size,
                              hipStream_t stream) {
    const float* x   = (const float*)d_in[0];   // (8,64,4096) f32
    const int*   qw  = (const int*)  d_in[1];   // (11008,256) i32
    const float* lut = (const float*)d_in[2];   // (16,2) f32
    float*       out = (float*)d_out;           // (512,11008) f32

    dim3 grid(OUT_FEATURES / TILE_M, BATCH / TILE_B);   // (43, 8)
    vq_wmma_gemm<<<grid, 256, 0, stream>>>(x, qw, lut, out);
}
